// MultiHeadAttention_74947179315838
// MI455X (gfx1250) — matrix operand order repaired and checked by reading, not compiled, not run on hardware
//
#include <hip/hip_runtime.h>
#include <hip/hip_bf16.h>

// Problem constants (B=2, S=2048, D=1024, H=16, HD=64)
#define BB 2
#define SS 2048
#define DD 1024
#define HH 16
#define HDIM 64
#define MM (BB * SS)

typedef __bf16 bf16;
typedef __attribute__((ext_vector_type(16))) __bf16 v16bf;
typedef __attribute__((ext_vector_type(8))) float v8f;
typedef __attribute__((ext_vector_type(4))) unsigned int v4u;
typedef __attribute__((ext_vector_type(4))) int v4i;
typedef __attribute__((ext_vector_type(8))) int v8i;

struct alignas(16) B16x8 { bf16 h[8]; };

__device__ inline v8f wmma_bf16(v16bf a, v16bf b, v8f c) {
  // (neg_a, A, neg_b, B, c_mod, C, reuse_a, reuse_b)
  return __builtin_amdgcn_wmma_f32_16x16x32_bf16(false, a, false, b, (short)0, c,
                                                 false, false);
}

// ---------------------------------------------------------------------------
// Tensor Data Mover: 2D bf16 tile load, global -> LDS, with row padding so the
// LDS layout lands on our bank-staggered stride.
//   pad_interval_code v: pad inserted every 2^(v+1) DWORDs; pad_amount = 1 DWORD.
//   tile0 = contiguous elements per row, tile1 = rows, stride0 = row stride (elems).
// D# packed per CDNA5 ISA 8.3/8.4 (group0: count|lds|global|type=2; group1:
// data_size=2B, pad fields, tensor/tile dims, dim0 stride). tensor_dim==tile_dim
// so the OOB check can never trigger regardless of origin interpretation.
// 6-arg builtin form (clang-23/therock): (g0, g1, g2, g3, g4, cpol).
// ---------------------------------------------------------------------------
__device__ inline void tdm_load_tile_bf16(unsigned lds_byte_off, const bf16* gptr,
                                          unsigned tile0, unsigned tile1,
                                          unsigned stride0_elems,
                                          unsigned pad_interval_code) {
  const unsigned long long ga = (unsigned long long)(uintptr_t)gptr;
  v4u g0;
  g0[0] = 1u;                                                 // count=1 (valid user D#)
  g0[1] = lds_byte_off;                                       // lds_addr (bytes)
  g0[2] = (unsigned)ga;                                       // global_addr[31:0]
  g0[3] = (unsigned)((ga >> 32) & 0x01FFFFFFu) | (2u << 30);  // global_addr[56:32] | type=2
  v8i g1;
  g1[0] = (int)((1u << 16) | (1u << 20) | (pad_interval_code << 22));
  g1[1] = (int)((tile0 & 0xFFFFu) << 16);                                // tensor_dim0 lo
  g1[2] = (int)(((tile0 >> 16) & 0xFFFFu) | ((tile1 & 0xFFFFu) << 16));  // td0 hi | td1 lo
  g1[3] = (int)(((tile1 >> 16) & 0xFFFFu) | ((tile0 & 0xFFFFu) << 16));  // td1 hi | tile_dim0
  g1[4] = (int)(tile1 & 0xFFFFu);                                        // tile_dim1 (tile_dim2=0)
  g1[5] = (int)stride0_elems;                                            // dim0 stride lo
  g1[6] = 0;                                                             // stride hi | dim1 stride
  g1[7] = 0;
  v4i z4;
  z4[0] = z4[1] = z4[2] = z4[3] = 0;
  v8i z8;
  z8[0] = z8[1] = z8[2] = z8[3] = z8[4] = z8[5] = z8[6] = z8[7] = 0;
  __builtin_amdgcn_tensor_load_to_lds(g0, g1, z4, z4, z8, 0);
}

// ---------------------------------------------------------------------------
// GEMM: Out[M][N] = A[M][K] @ W[N][K]^T   (torch Linear convention)
// A fp32 (convert on stage) or bf16 (TDM-DMA'd straight into padded LDS);
// W fp32 -> bf16 on stage; Out bf16 or fp32.
// Block = 256 threads (8 wave32), tile = 128x128, K step = 32.
// ---------------------------------------------------------------------------
template <bool A_BF16, bool OUT_F32>
__global__ __launch_bounds__(256) void gemm_xwT(const void* __restrict__ Aptr,
                                                const float* __restrict__ W,
                                                void* __restrict__ Out,
                                                int M, int N, int K) {
  constexpr int KP = 34;  // 32 elems + 1 DWORD pad (matches TDM pad_interval=3)
  __shared__ bf16 As[128][KP];
  __shared__ bf16 Bs[128][KP];  // Bs[n][k] = W[n0+n][k0+k]

  const int t = threadIdx.x;
  const int wave = t >> 5, lane = t & 31;
  const int half = lane >> 4, l16 = lane & 15;
  const int m0 = blockIdx.y * 128, n0 = blockIdx.x * 128;
  const int srow = t >> 1, kseg = (t & 1) * 16;

  v8f acc[8];
#pragma unroll
  for (int i = 0; i < 8; ++i)
#pragma unroll
    for (int j = 0; j < 8; ++j) acc[i][j] = 0.0f;

  for (int k0 = 0; k0 < K; k0 += 32) {
    __syncthreads();
    // ---- stage A tile (128x32) ----
    if constexpr (A_BF16) {
      if (t < 32) {
        const bf16* A = (const bf16*)Aptr;
        tdm_load_tile_bf16((unsigned)(uintptr_t)&As[0][0],
                           A + (size_t)m0 * K + k0, 32, 128, (unsigned)K, 3);
      }
    } else {
      const float* A = (const float*)Aptr;
      const float4* p =
          reinterpret_cast<const float4*>(A + (size_t)(m0 + srow) * K + k0 + kseg);
#pragma unroll
      for (int q = 0; q < 4; ++q) {
        float4 f = p[q];
        As[srow][kseg + 4 * q + 0] = (bf16)f.x;
        As[srow][kseg + 4 * q + 1] = (bf16)f.y;
        As[srow][kseg + 4 * q + 2] = (bf16)f.z;
        As[srow][kseg + 4 * q + 3] = (bf16)f.w;
      }
      if (k0 + 32 < K)
        __builtin_prefetch(A + (size_t)(m0 + srow) * K + k0 + 32 + kseg, 0, 3);
    }
    // ---- stage W tile (128 n-rows x 32 k), untransposed ----
    {
      const float4* p =
          reinterpret_cast<const float4*>(W + (size_t)(n0 + srow) * K + k0 + kseg);
#pragma unroll
      for (int q = 0; q < 4; ++q) {
        float4 f = p[q];
        Bs[srow][kseg + 4 * q + 0] = (bf16)f.x;
        Bs[srow][kseg + 4 * q + 1] = (bf16)f.y;
        Bs[srow][kseg + 4 * q + 2] = (bf16)f.z;
        Bs[srow][kseg + 4 * q + 3] = (bf16)f.w;
      }
      if (k0 + 32 < K)
        __builtin_prefetch(W + (size_t)(n0 + srow) * K + k0 + 32 + kseg, 0, 3);
    }
    if constexpr (A_BF16) {
      if (t < 32) __builtin_amdgcn_s_wait_tensorcnt(0);
    }
    __syncthreads();

    // ---- A fragment: wave's 16 rows, K window of 32 ----
    v16bf afrag;
    {
      const int row = wave * 16 + l16;
#pragma unroll
      for (int i = 0; i < 8; ++i) {
        afrag[i] = As[row][half * 16 + i];
        afrag[8 + i] = As[row][half * 16 + 8 + i];
      }
    }
    // ---- 8 N-subtiles of 16 ----
#pragma unroll
    for (int nt = 0; nt < 8; ++nt) {
      v16bf bfrag;
      const int col = nt * 16 + l16;
      const int kh = half * 16;
#pragma unroll
      for (int i = 0; i < 16; ++i) bfrag[i] = Bs[col][kh + i];
      acc[nt] = wmma_bf16(afrag, bfrag, acc[nt]);
    }
  }

  // ---- store C (f32 layout: VGPR r -> row half*8+r, col = lane&15) ----
#pragma unroll
  for (int nt = 0; nt < 8; ++nt) {
#pragma unroll
    for (int r = 0; r < 8; ++r) {
      const int row = m0 + wave * 16 + half * 8 + r;
      const int col = n0 + nt * 16 + l16;
      const float v = acc[nt][r];
      if constexpr (OUT_F32)
        ((float*)Out)[(size_t)row * N + col] = v;
      else
        ((bf16*)Out)[(size_t)row * N + col] = (bf16)v;
    }
  }
}

// ---------------------------------------------------------------------------
// Flash attention: one block = 128 query rows of one (batch, head).
// Double-buffered 64-key tiles: K DMA'd by the TDM (async, TENSORcnt),
// V vector-staged transposed; online softmax in exp2 domain.
// ---------------------------------------------------------------------------
__global__ __launch_bounds__(256) void flash_attn(const bf16* __restrict__ Q,
                                                  const bf16* __restrict__ Kg,
                                                  const bf16* __restrict__ V,
                                                  bf16* __restrict__ Ctx) {
  constexpr int KP = 66;       // 64 elems + 1 DWORD pad (matches TDM pad_interval=4)
  constexpr int NKB = SS / 64;
  __shared__ bf16 Ks[2][64][KP];     // Ks[buf][key][d]   (TDM destination)
  __shared__ bf16 Vs[2][64][KP];     // Vs[buf][d][key]   (transposed on stage)
  __shared__ bf16 Ps[8][16][KP];     // per-wave P re-layout buffer

  const int t = threadIdx.x;
  const int wave = t >> 5, lane = t & 31;
  const int half = lane >> 4, l16 = lane & 15;
  const int bh = blockIdx.y;
  const int bidx = bh / HH, h = bh % HH;
  const int q0 = blockIdx.x * 128 + wave * 16;
  const size_t base = (size_t)bidx * SS * DD + (size_t)h * HDIM;

  const float qscale = 0.125f * 1.44269504088896340736f;  // 1/sqrt(64) * log2(e)

  // Q fragments (2 K-windows of 32 covering HD=64), scale folded in
  v16bf aq[2];
  {
    const bf16* qp = Q + base + (size_t)(q0 + l16) * DD;
#pragma unroll
    for (int kw = 0; kw < 2; ++kw)
#pragma unroll
      for (int i = 0; i < 8; ++i) {
        aq[kw][i] = (bf16)((float)qp[kw * 32 + half * 16 + i] * qscale);
        aq[kw][8 + i] = (bf16)((float)qp[kw * 32 + half * 16 + 8 + i] * qscale);
      }
  }

  float mrow[8], lrow[8];
#pragma unroll
  for (int r = 0; r < 8; ++r) { mrow[r] = -1e30f; lrow[r] = 0.0f; }
  v8f o[4];
#pragma unroll
  for (int d = 0; d < 4; ++d)
#pragma unroll
    for (int r = 0; r < 8; ++r) o[d][r] = 0.0f;

  const int srow = t >> 2, sseg = (t & 3) * 16;  // 64 rows x four 16-wide segs

  auto issueK = [&](int kb, int buf) {
    if (t < 32)
      tdm_load_tile_bf16((unsigned)(uintptr_t)&Ks[buf][0][0],
                         Kg + base + (size_t)kb * DD, 64, 64, DD, 4);
  };
  auto stageV = [&](int kb, int buf) {
    const B16x8* vp = reinterpret_cast<const B16x8*>(
        V + base + (size_t)(kb + srow) * DD + sseg);
    B16x8 v0 = vp[0], v1 = vp[1];
#pragma unroll
    for (int i = 0; i < 8; ++i) {
      Vs[buf][sseg + i][srow] = v0.h[i];
      Vs[buf][sseg + 8 + i][srow] = v1.h[i];
    }
  };

  // ---- prologue: tile 0 ----
  issueK(0, 0);
  stageV(0, 0);
  if (t < 32) __builtin_amdgcn_s_wait_tensorcnt(0);
  __syncthreads();

  for (int it = 0; it < NKB; ++it) {
    const int cur = it & 1;
    // prefetch next tile into the other buffer while we compute on this one
    if (it + 1 < NKB) {
      issueK((it + 1) * 64, cur ^ 1);
      stageV((it + 1) * 64, cur ^ 1);
    }

    // ---- scores: 4 key-subtiles x 2 WMMA (d = 0..31, 32..63) ----
    v8f s4[4];
#pragma unroll
    for (int kt = 0; kt < 4; ++kt) {
      v8f s;
#pragma unroll
      for (int r = 0; r < 8; ++r) s[r] = 0.0f;
      const int kk = kt * 16 + l16;
#pragma unroll
      for (int kw = 0; kw < 2; ++kw) {
        v16bf bfrag;
        const int dh = kw * 32 + half * 16;
#pragma unroll
        for (int i = 0; i < 16; ++i) bfrag[i] = Ks[cur][kk][dh + i];
        s = wmma_bf16(aq[kw], bfrag, s);
      }
      s4[kt] = s;
    }

    // ---- online softmax (base-2 domain) ----
    float alpha[8];
#pragma unroll
    for (int r = 0; r < 8; ++r) {
      float mx = s4[0][r];
      mx = fmaxf(mx, s4[1][r]);
      mx = fmaxf(mx, s4[2][r]);
      mx = fmaxf(mx, s4[3][r]);
#pragma unroll
      for (int off = 1; off < 16; off <<= 1) mx = fmaxf(mx, __shfl_xor(mx, off, 32));
      const float mnew = fmaxf(mrow[r], mx);
      alpha[r] = exp2f(mrow[r] - mnew);
      float sum = 0.0f;
#pragma unroll
      for (int kt = 0; kt < 4; ++kt) {
        const float p = exp2f(s4[kt][r] - mnew);
        s4[kt][r] = p;
        sum += p;
      }
#pragma unroll
      for (int off = 1; off < 16; off <<= 1) sum += __shfl_xor(sum, off, 32);
      lrow[r] = lrow[r] * alpha[r] + sum;
      mrow[r] = mnew;
    }
#pragma unroll
    for (int d = 0; d < 4; ++d)
#pragma unroll
      for (int r = 0; r < 8; ++r) o[d][r] *= alpha[r];

    // ---- re-layout P (C-format -> A-format) through per-wave LDS ----
#pragma unroll
    for (int kt = 0; kt < 4; ++kt)
#pragma unroll
      for (int r = 0; r < 8; ++r)
        Ps[wave][half * 8 + r][kt * 16 + l16] = (bf16)s4[kt][r];
    // within-wave LDS RAW: DS ops are in-order per wave; make it explicit
    asm volatile("s_wait_dscnt 0" ::: "memory");

    v16bf ap[2];
#pragma unroll
    for (int kw = 0; kw < 2; ++kw)
#pragma unroll
      for (int i = 0; i < 8; ++i) {
        ap[kw][i] = Ps[wave][l16][kw * 32 + half * 16 + i];
        ap[kw][8 + i] = Ps[wave][l16][kw * 32 + half * 16 + 8 + i];
      }

    // ---- O += P @ V : 4 d-subtiles x 2 WMMA (keys 0..31, 32..63) ----
#pragma unroll
    for (int dt = 0; dt < 4; ++dt) {
      const int dcol = dt * 16 + l16;
#pragma unroll
      for (int kw = 0; kw < 2; ++kw) {
        v16bf bfrag;
        const int kh = kw * 32 + half * 16;
#pragma unroll
        for (int i = 0; i < 16; ++i) bfrag[i] = Vs[cur][dcol][kh + i];
        o[dt] = wmma_bf16(ap[kw], bfrag, o[dt]);
      }
    }

    // publish the prefetched buffer (TDM drained by wave0 before the barrier)
    if (it + 1 < NKB) {
      if (t < 32) __builtin_amdgcn_s_wait_tensorcnt(0);
      __syncthreads();
    }
  }

  // ---- normalize and write ctx (bf16) ----
#pragma unroll
  for (int r = 0; r < 8; ++r) {
    const float inv = 1.0f / lrow[r];
    bf16* cp = Ctx + base + (size_t)(q0 + half * 8 + r) * DD;
#pragma unroll
    for (int dt = 0; dt < 4; ++dt) cp[dt * 16 + l16] = (bf16)(o[dt][r] * inv);
  }
}

// ---------------------------------------------------------------------------
extern "C" void kernel_launch(void* const* d_in, const int* in_sizes, int n_in,
                              void* d_out, int out_size, void* d_ws, size_t ws_size,
                              hipStream_t stream) {
  (void)in_sizes; (void)n_in; (void)out_size; (void)ws_size;
  const float* queries = (const float*)d_in[0];
  const float* keys    = (const float*)d_in[1];
  const float* values  = (const float*)d_in[2];
  const float* Wq      = (const float*)d_in[3];
  const float* Wk      = (const float*)d_in[4];
  const float* Wv      = (const float*)d_in[5];
  const float* Wo      = (const float*)d_in[6];
  float* out = (float*)d_out;

  const size_t MD = (size_t)MM * DD;
  bf16* Qb = (bf16*)d_ws;  // 8 MB each; 32 MB total — lives in L2
  bf16* Kb = Qb + MD;
  bf16* Vb = Kb + MD;
  bf16* Cx = Vb + MD;

  dim3 blk(256);
  dim3 gp(DD / 128, MM / 128);  // 8 x 32 blocks

  gemm_xwT<false, false><<<gp, blk, 0, stream>>>(queries, Wq, Qb, MM, DD, DD);
  gemm_xwT<false, false><<<gp, blk, 0, stream>>>(keys,    Wk, Kb, MM, DD, DD);
  gemm_xwT<false, false><<<gp, blk, 0, stream>>>(values,  Wv, Vb, MM, DD, DD);

  dim3 ga(SS / 128, BB * HH);  // 16 x 32 blocks
  flash_attn<<<ga, blk, 0, stream>>>(Qb, Kb, Vb, Cx);

  gemm_xwT<true, true><<<gp, blk, 0, stream>>>(Cx, Wo, out, MM, DD, DD);
}
